// AttentionHead_19688130085013
// MI455X (gfx1250) — matrix operand order repaired and checked by reading, not compiled, not run on hardware
//
#include <hip/hip_runtime.h>
#include <hip/hip_bf16.h>
#include <math.h>

// ---------------------------------------------------------------------------
// CDNA5 (gfx1250) attention block:
//   pre-convert to bf16 -> QKV GEMM (WMMA, async-LDS double-buffered) -> RoPE
//   -> causal flash attention (WMMA) -> proj GEMM (WMMA).
// ---------------------------------------------------------------------------

typedef __bf16 bf16;
typedef bf16  v16bf __attribute__((ext_vector_type(16)));
typedef float v8f   __attribute__((ext_vector_type(8)));
typedef int   v4i   __attribute__((ext_vector_type(4)));

#define NUM_HEADS 32
#define EMB       4096
#define HEAD_DIM  128
#define BATCH     2
#define SEQ       2048
#define QKV_LD    (3 * EMB)

// ----- gfx1250 async global->LDS path (guarded: falls back to ds stores) ---
#if __has_builtin(__builtin_amdgcn_global_load_async_to_lds_b128)
#define HAVE_ASYNC 1
typedef __attribute__((address_space(1))) v4i* gbl_v4i_p;
typedef __attribute__((address_space(3))) v4i* lds_v4i_p;
#define ASYNC_LD_B128(gsrc, ldst)                                            \
  __builtin_amdgcn_global_load_async_to_lds_b128(                            \
      (gbl_v4i_p)(gsrc), (lds_v4i_p)(ldst), 0, 0)
#endif

static __device__ __forceinline__ void wait_async_all() {
#if defined(HAVE_ASYNC)
#if __has_builtin(__builtin_amdgcn_s_wait_asynccnt)
  __builtin_amdgcn_s_wait_asynccnt(0);
#else
  asm volatile("s_wait_asynccnt 0" ::: "memory");
#endif
#endif
}

static __device__ __forceinline__ v8f wmma_bf16(v16bf a, v16bf b, v8f c) {
  return __builtin_amdgcn_wmma_f32_16x16x32_bf16(
      /*neg_a=*/false, a, /*neg_b=*/false, b,
      /*c_mod=*/(short)0, c, /*reuse_a=*/false, /*reuse_b=*/false);
}

static __device__ __forceinline__ v8f zero8() {
  v8f z;
#pragma unroll
  for (int i = 0; i < 8; ++i) z[i] = 0.0f;
  return z;
}

// K index of VGPR pair v in a bf16 16x32 A-fragment (ISA 7.12.2, wave32):
// lanes 0-15 -> K {0..7,16..23}; lanes 16-31 -> K {8..15,24..31}.
static __device__ __forceinline__ int a_kidx(int v, int half16) {
  return ((v < 4) ? 0 : 16) + half16 * 8 + (v & 3) * 2;
}

// ---------------------------------------------------------------------------
// f32 -> bf16 elementwise convert (for x).
// ---------------------------------------------------------------------------
__global__ __launch_bounds__(256)
void convert_bf16(const float* __restrict__ in, bf16* __restrict__ out) {
  int i = blockIdx.x * 256 + threadIdx.x;
  out[i] = (bf16)in[i];
}

// ---------------------------------------------------------------------------
// Tiled transpose + convert: in[R][C] f32  ->  out[C][R] bf16.
// Block = 256 threads handling a 32x32 tile.
// ---------------------------------------------------------------------------
__global__ __launch_bounds__(256)
void transpose_to_bf16(const float* __restrict__ in, bf16* __restrict__ out,
                       int R, int C) {
  __shared__ float tile[32][33];
  int c0 = blockIdx.x * 32;
  int r0 = blockIdx.y * 32;
  int tx = threadIdx.x & 31;
  int ty = threadIdx.x >> 5;   // 0..7
#pragma unroll
  for (int i = 0; i < 4; ++i) {
    int r = ty + i * 8;
    tile[r][tx] = in[(size_t)(r0 + r) * C + (c0 + tx)];
  }
  __syncthreads();
#pragma unroll
  for (int i = 0; i < 4; ++i) {
    int r = ty + i * 8;
    out[(size_t)(c0 + r) * R + (r0 + tx)] = (bf16)tile[tx][r];
  }
}

// ---------------------------------------------------------------------------
// bf16 WMMA GEMM: C[M,N](f32) = A[M,K](bf16) * Bt[N,K](bf16, pre-transposed)
// Block tile 128x128, 8 waves (4x2), wave tile 32x64 (2x4 WMMA tiles).
// Double-buffered LDS staging; async global->LDS when available.
// ---------------------------------------------------------------------------
__global__ __launch_bounds__(256)
void gemm_bf16_wmma(const bf16* __restrict__ A, const bf16* __restrict__ Bt,
                    float* __restrict__ C, int M, int N, int K) {
  const int LDSS = 40;   // bf16 row stride: 80B, 16B-aligned, bank-spread pad
  __shared__ __align__(16) bf16 As[2][128 * 40];
  __shared__ __align__(16) bf16 Bs[2][128 * 40];

  const int tid    = threadIdx.x;
  const int lane   = tid & 31;
  const int wave   = tid >> 5;
  const int l16    = lane & 15;
  const int half16 = (lane >> 4) & 1;
  const int wm     = wave >> 1;        // 0..3 -> 32-row strip
  const int wn     = wave & 1;         // 0..1 -> 64-col strip
  const int m0     = blockIdx.x * 128;
  const int n0     = blockIdx.y * 128;

  // staging assignment: 512 16B-chunks per tile per matrix, 2 each per thread
  int srow[2], sq[2];
#pragma unroll
  for (int i = 0; i < 2; ++i) {
    int chunk = tid * 2 + i;
    srow[i] = chunk >> 2;              // 0..127
    sq[i]   = (chunk & 3) * 8;         // bf16 offset {0,8,16,24}
  }

  auto stage = [&](int p, int kk0) {
#pragma unroll
    for (int i = 0; i < 2; ++i) {
      const bf16* ga = A  + (size_t)(m0 + srow[i]) * K + kk0 + sq[i];
      const bf16* gb = Bt + (size_t)(n0 + srow[i]) * K + kk0 + sq[i];
#if defined(HAVE_ASYNC)
      ASYNC_LD_B128(ga, &As[p][srow[i] * LDSS + sq[i]]);
      ASYNC_LD_B128(gb, &Bs[p][srow[i] * LDSS + sq[i]]);
#else
      *(v4i*)&As[p][srow[i] * LDSS + sq[i]] = *(const v4i*)ga;
      *(v4i*)&Bs[p][srow[i] * LDSS + sq[i]] = *(const v4i*)gb;
#endif
    }
  };

  v8f acc[2][4];
#pragma unroll
  for (int tm = 0; tm < 2; ++tm)
#pragma unroll
    for (int tn = 0; tn < 4; ++tn) acc[tm][tn] = zero8();

  // prologue: stage first k-tile
  stage(0, 0);
  wait_async_all();
  __syncthreads();

  int p = 0;
  for (int k0 = 0; k0 < K; k0 += 32, p ^= 1) {
    if (k0 + 32 < K) {
      stage(p ^ 1, k0 + 32);   // overlap next-tile copy with this tile's math
      // L2 prefetch two tiles ahead
      if (k0 + 64 < K) {
        __builtin_prefetch(A  + (size_t)(m0 + srow[0]) * K + k0 + 64 + sq[0], 0, 1);
        __builtin_prefetch(Bt + (size_t)(n0 + srow[0]) * K + k0 + 64 + sq[0], 0, 1);
      }
    }

    v16bf afrag[2], bfrag[4];
#pragma unroll
    for (int tm = 0; tm < 2; ++tm) {
      int r = wm * 32 + tm * 16 + l16;
#pragma unroll
      for (int v = 0; v < 8; ++v) {
        int kk = a_kidx(v, half16);
        afrag[tm][2 * v]     = As[p][r * LDSS + kk];
        afrag[tm][2 * v + 1] = As[p][r * LDSS + kk + 1];
      }
    }
#pragma unroll
    for (int tn = 0; tn < 4; ++tn) {
      int c = wn * 64 + tn * 16 + l16;
#pragma unroll
      for (int v = 0; v < 8; ++v) {
        int kk = a_kidx(v, half16);
        bfrag[tn][2 * v]     = Bs[p][c * LDSS + kk];
        bfrag[tn][2 * v + 1] = Bs[p][c * LDSS + kk + 1];
      }
    }
#pragma unroll
    for (int tm = 0; tm < 2; ++tm)
#pragma unroll
      for (int tn = 0; tn < 4; ++tn)
        acc[tm][tn] = wmma_bf16(afrag[tm], bfrag[tn], acc[tm][tn]);

    wait_async_all();    // next tile landed (issued before the 8 WMMAs)
    __syncthreads();
  }

  // epilogue: C-fragment row = 8*half16 + v, col = l16
#pragma unroll
  for (int tm = 0; tm < 2; ++tm)
#pragma unroll
    for (int tn = 0; tn < 4; ++tn)
#pragma unroll
      for (int v = 0; v < 8; ++v) {
        int row = m0 + wm * 32 + tm * 16 + half16 * 8 + v;
        int col = n0 + wn * 64 + tn * 16 + l16;
        C[(size_t)row * N + col] = acc[tm][tn][v];
      }
}

// ---------------------------------------------------------------------------
// RoPE applied in-place to q,k halves of the qkv workspace (f32).
// ---------------------------------------------------------------------------
__global__ __launch_bounds__(256)
void rope_kernel(float* __restrict__ qkv, const float* __restrict__ fcos,
                 const float* __restrict__ fsin) {
  int tid = blockIdx.x * blockDim.x + threadIdx.x;   // < 2^23
  int i = tid & 63;
  int h = (tid >> 6) & 31;
  int s = (tid >> 11) & 2047;
  int b = tid >> 22;
  float c  = fcos[s * 64 + i];
  float sn = fsin[s * 64 + i];
  float* q = qkv + ((size_t)(b * SEQ + s)) * QKV_LD + h * HEAD_DIM + 2 * i;
  float x0 = q[0], x1 = q[1];
  q[0] = x0 * c - x1 * sn;
  q[1] = x0 * sn + x1 * c;
  float* k = q + EMB;
  x0 = k[0]; x1 = k[1];
  k[0] = x0 * c - x1 * sn;
  k[1] = x0 * sn + x1 * c;
}

// ---------------------------------------------------------------------------
// Flash attention (causal, streaming softmax). One wave = one 16-row Q tile.
// Emits bf16 output directly (input of the projection GEMM).
// ---------------------------------------------------------------------------
__global__ __launch_bounds__(256)
void flash_attn_wmma(const float* __restrict__ qkv, bf16* __restrict__ out) {
  __shared__ float Plds[8][16 * 33];   // per-wave C->A transpose staging

  const int tid    = threadIdx.x;
  const int lane   = tid & 31;
  const int wave   = tid >> 5;
  const int l16    = lane & 15;
  const int half16 = (lane >> 4) & 1;

  const int qblk = blockIdx.x % (SEQ / 128);
  const int bh   = blockIdx.x / (SEQ / 128);
  const int h    = bh % NUM_HEADS;
  const int b    = bh / NUM_HEADS;
  const int q0   = qblk * 128 + wave * 16;

  const float* qbase = qkv + ((size_t)b * SEQ) * QKV_LD + h * HEAD_DIM;
  const float* kbase = qbase + EMB;
  const float* vbase = qbase + 2 * EMB;
  const float  scale = 0.08838834764831845f;   // 1/sqrt(128)

  // Q fragments: 16 rows x 128 dims = 4 A-fragments of K=32
  v16bf qfrag[4];
  {
    int qrow = q0 + l16;
#pragma unroll
    for (int dc = 0; dc < 4; ++dc)
#pragma unroll
      for (int v = 0; v < 8; ++v) {
        int d = dc * 32 + a_kidx(v, half16);
        const float* p = qbase + (size_t)qrow * QKV_LD + d;
        qfrag[dc][2 * v]     = (bf16)p[0];
        qfrag[dc][2 * v + 1] = (bf16)p[1];
      }
  }

  v8f o[8];
#pragma unroll
  for (int t = 0; t < 8; ++t) o[t] = zero8();
  float mrow[8], lrow[8];
#pragma unroll
  for (int v = 0; v < 8; ++v) { mrow[v] = -3.0e38f; lrow[v] = 0.0f; }

  for (int j0 = 0; j0 <= q0 + 15; j0 += 32) {      // causal chunk bound
    // ---- scores: two 16x16 tiles over 32 keys ----
    v8f s[2];
#pragma unroll
    for (int t = 0; t < 2; ++t) {
      s[t] = zero8();
      int key = j0 + t * 16 + l16;
#pragma unroll
      for (int dc = 0; dc < 4; ++dc) {
        v16bf kf;   // B-fragment of K^T: (K=32 dims) x (N=16 keys)
#pragma unroll
        for (int v = 0; v < 8; ++v) {
          int d = dc * 32 + a_kidx(v, half16);
          const float* p = kbase + (size_t)key * QKV_LD + d;
          kf[2 * v]     = (bf16)p[0];
          kf[2 * v + 1] = (bf16)p[1];
        }
        s[t] = wmma_bf16(qfrag[dc], kf, s[t]);
      }
    }
    // ---- scale + causal mask (row = q0 + 8*half16 + v, col = key) ----
    int rbase = q0 + half16 * 8;
#pragma unroll
    for (int t = 0; t < 2; ++t)
#pragma unroll
      for (int v = 0; v < 8; ++v) {
        int key = j0 + t * 16 + l16;
        float val = s[t][v] * scale;
        s[t][v] = (key > rbase + v) ? -3.0e38f : val;
      }
    // ---- streaming softmax update ----
#pragma unroll
    for (int v = 0; v < 8; ++v) {
      float mx = fmaxf(s[0][v], s[1][v]);
#pragma unroll
      for (int off = 1; off < 16; off <<= 1)
        mx = fmaxf(mx, __shfl_xor(mx, off, 32));   // stays inside 16-lane half
      float mnew = fmaxf(mrow[v], mx);
      float p0 = __expf(s[0][v] - mnew);
      float p1 = __expf(s[1][v] - mnew);
      s[0][v] = p0; s[1][v] = p1;
      float ps = p0 + p1;
#pragma unroll
      for (int off = 1; off < 16; off <<= 1)
        ps += __shfl_xor(ps, off, 32);
      float corr = __expf(mrow[v] - mnew);
      lrow[v] = lrow[v] * corr + ps;
      mrow[v] = mnew;
#pragma unroll
      for (int t2 = 0; t2 < 8; ++t2) o[t2][v] *= corr;
    }
    // ---- transpose P (C-layout) -> A-fragment via per-wave LDS ----
#pragma unroll
    for (int t = 0; t < 2; ++t)
#pragma unroll
      for (int v = 0; v < 8; ++v)
        Plds[wave][(half16 * 8 + v) * 33 + t * 16 + l16] = s[t][v];
    __builtin_amdgcn_wave_barrier();   // DS ops in-order within wave
    v16bf pfrag;
#pragma unroll
    for (int v = 0; v < 8; ++v) {
      int kk = a_kidx(v, half16);
      pfrag[2 * v]     = (bf16)Plds[wave][l16 * 33 + kk];
      pfrag[2 * v + 1] = (bf16)Plds[wave][l16 * 33 + kk + 1];
    }
    // ---- O += P @ V over this 32-key chunk (8 output dim-tiles) ----
#pragma unroll
    for (int t2 = 0; t2 < 8; ++t2) {
      v16bf vf;   // B-fragment: (K=32 keys) x (N=16 dims)
      int dcol = t2 * 16 + l16;
#pragma unroll
      for (int v = 0; v < 8; ++v) {
        int key = j0 + a_kidx(v, half16);
        vf[2 * v]     = (bf16)vbase[(size_t)key * QKV_LD + dcol];
        vf[2 * v + 1] = (bf16)vbase[(size_t)(key + 1) * QKV_LD + dcol];
      }
      o[t2] = wmma_bf16(pfrag, vf, o[t2]);
    }
  }

  // ---- normalize + write bf16 (B, S, EMB) ----
  bf16* obase = out + ((size_t)b * SEQ + q0) * EMB + h * HEAD_DIM;
#pragma unroll
  for (int t2 = 0; t2 < 8; ++t2)
#pragma unroll
    for (int v = 0; v < 8; ++v) {
      int row = half16 * 8 + v;
      int col = t2 * 16 + l16;
      obase[(size_t)row * EMB + col] = (bf16)(o[t2][v] / lrow[v]);
    }
}

// ---------------------------------------------------------------------------
extern "C" void kernel_launch(void* const* d_in, const int* in_sizes, int n_in,
                              void* d_out, int out_size, void* d_ws, size_t ws_size,
                              hipStream_t stream) {
  (void)in_sizes; (void)n_in; (void)out_size; (void)ws_size;
  const float* x       = (const float*)d_in[0];
  const float* w_atten = (const float*)d_in[1];
  const float* w_proj  = (const float*)d_in[2];
  const float* fcos    = (const float*)d_in[3];
  const float* fsin    = (const float*)d_in[4];
  float* out = (float*)d_out;

  const int M = BATCH * SEQ;   // 4096

  // workspace carve-up (all 16B-aligned)
  char* ws = (char*)d_ws;
  float* qkv  = (float*)ws;                 ws += (size_t)M * QKV_LD * 4;      // 201.3 MB
  bf16* x_bf  = (bf16*)ws;                  ws += (size_t)M * EMB * 2;         //  33.6 MB
  bf16* wat_t = (bf16*)ws;                  ws += (size_t)(3 * EMB) * EMB * 2; // 100.7 MB
  bf16* wpj_t = (bf16*)ws;                  ws += (size_t)EMB * EMB * 2;       //  33.6 MB
  bf16* attn  = (bf16*)ws;                                                    //  33.6 MB

  dim3 blk(256);

  // 0) one-shot precision/layout conversion (memory-bound)
  convert_bf16<<<(M * EMB) / 256, blk, 0, stream>>>(x, x_bf);
  transpose_to_bf16<<<dim3((3 * EMB) / 32, EMB / 32), blk, 0, stream>>>(
      w_atten, wat_t, EMB, 3 * EMB);
  transpose_to_bf16<<<dim3(EMB / 32, EMB / 32), blk, 0, stream>>>(
      w_proj, wpj_t, EMB, EMB);

  // 1) qkv = x @ w_atten   (4096 x 12288 x 4096)
  gemm_bf16_wmma<<<dim3(M / 128, (3 * EMB) / 128), blk, 0, stream>>>(
      x_bf, wat_t, qkv, M, 3 * EMB, EMB);

  // 2) RoPE on q,k in place
  rope_kernel<<<(BATCH * SEQ * NUM_HEADS * (HEAD_DIM / 2)) / 256, blk, 0,
                stream>>>(qkv, fcos, fsin);

  // 3) causal flash attention -> attn (bf16, B,S,E)
  flash_attn_wmma<<<BATCH * NUM_HEADS * (SEQ / 128), blk, 0, stream>>>(qkv,
                                                                        attn);

  // 4) out = attn @ w_proj  (4096 x 4096 x 4096)
  gemm_bf16_wmma<<<dim3(M / 128, EMB / 128), blk, 0, stream>>>(
      attn, wpj_t, out, M, EMB, EMB);
}
